// DynamicConv2D_6957847020100
// MI455X (gfx1250) — matrix operand order repaired and checked by reading, not compiled, not run on hardware
//
#include <hip/hip_runtime.h>
#include <hip/hip_fp16.h>

#define B_    32
#define CIN   256
#define COUT  256
#define HH    56
#define WW    56
#define HW    3136
#define HP    58      // padded H/W
#define KEXP  4

typedef __attribute__((ext_vector_type(16))) _Float16 v16h;
typedef __attribute__((ext_vector_type(8)))  _Float16 v8h;
typedef __attribute__((ext_vector_type(8)))  float    v8f;

// ---------------- Phase 1: global average pool ----------------
__global__ __launch_bounds__(256) void pool_kernel(const float* __restrict__ x,
                                                   float* __restrict__ pooled) {
  __shared__ float red[256];
  const int bc = blockIdx.x;                       // b*CIN + c
  const float* p = x + (size_t)bc * HW;
  float s = 0.f;
  for (int i = threadIdx.x; i < HW; i += 256) s += p[i];
  red[threadIdx.x] = s;
  __syncthreads();
  for (int off = 128; off > 0; off >>= 1) {
    if ((int)threadIdx.x < off) red[threadIdx.x] += red[threadIdx.x + off];
    __syncthreads();
  }
  if (threadIdx.x == 0) pooled[bc] = red[0] * (1.0f / HW);
}

// ---------------- Phase 2: attention (fc1-relu-fc2-softmax), 1 block x 128 thr ----
__global__ __launch_bounds__(128) void attention_kernel(
    const float* __restrict__ pooled, const float* __restrict__ fc1_w,
    const float* __restrict__ fc1_b,  const float* __restrict__ fc2_w,
    const float* __restrict__ fc2_b,  float* __restrict__ att) {
  __shared__ float aS[B_][KEXP];
  __shared__ float lS[B_][KEXP];
  const int t = threadIdx.x;
  const int b = t >> 2, k = t & 3;
  float s = fc1_b[k];
  const float* pv = pooled + b * CIN;
  const float* wv = fc1_w + k * CIN;
  for (int c = 0; c < CIN; ++c) s += pv[c] * wv[c];
  aS[b][k] = fmaxf(s, 0.f);
  __syncthreads();
  float l = fc2_b[k];
  for (int j = 0; j < KEXP; ++j) l += aS[b][j] * fc2_w[k * KEXP + j];
  lS[b][k] = l;
  __syncthreads();
  float m = lS[b][0];
  for (int j = 1; j < KEXP; ++j) m = fmaxf(m, lS[b][j]);
  float d = 0.f;
  for (int j = 0; j < KEXP; ++j) d += __expf(lS[b][j] - m);
  att[t] = __expf(l - m) / d;
}

// ---------------- Phase 3: aggregate expert weights -> f16, layout [b][tap][cout][cin]
__global__ __launch_bounds__(256) void aggw_kernel(const float* __restrict__ w,
                                                   const float* __restrict__ att,
                                                   _Float16* __restrict__ aggwT) {
  const int b = blockIdx.y;
  const int f = blockIdx.x * 256 + threadIdx.x;    // 0 .. 9*256*256-1
  const int tap = f >> 16;
  const int rem = f & 0xFFFF;
  const int co  = rem >> 8;
  const int ci  = rem & 255;
  const float a0 = att[b * 4 + 0], a1 = att[b * 4 + 1];
  const float a2 = att[b * 4 + 2], a3 = att[b * 4 + 3];
  const size_t base = ((size_t)co * CIN + ci) * 9 + tap;   // weight[K][COUT][CIN][3][3]
  const size_t es = (size_t)COUT * CIN * 9;
  const float s = a0 * w[base] + a1 * w[base + es] + a2 * w[base + 2 * es] + a3 * w[base + 3 * es];
  aggwT[((size_t)((b * 9 + tap) * COUT + co)) * CIN + ci] = (_Float16)s;
}

// ---------------- Phase 4: aggregate bias ----------------
__global__ __launch_bounds__(256) void aggb_kernel(const float* __restrict__ bias,
                                                   const float* __restrict__ att,
                                                   float* __restrict__ aggb) {
  const int b = blockIdx.x, c = threadIdx.x;
  float s = 0.f;
  for (int k = 0; k < KEXP; ++k) s += att[b * 4 + k] * bias[k * COUT + c];
  aggb[b * COUT + c] = s;
}

// ---------------- Phase 5a: zero the padded NHWC image (vector stores) ----------
__global__ __launch_bounds__(256) void zero_kernel(v8h* __restrict__ p, int n8) {
  const v8h z = {};
  for (int i = blockIdx.x * blockDim.x + threadIdx.x; i < n8; i += gridDim.x * blockDim.x)
    p[i] = z;
}

// ---------------- Phase 5b: NCHW f32 -> padded NHWC f16 transpose -----------------
// One block per (h, b): stage the 256x56 plane in LDS (coalesced reads),
// write 56x256 cin-contiguous f16 (coalesced writes). Stride 57 kills conflicts.
__global__ __launch_bounds__(256) void pad_transpose_kernel(const float* __restrict__ x,
                                                            _Float16* __restrict__ xpad) {
  __shared__ float plane[CIN * 57];
  const int h = blockIdx.x;            // 0..55
  const int b = blockIdx.y;            // 0..31
  const int t = threadIdx.x;
  // load: x[b][ci][h][w], w-contiguous
  for (int i = t; i < CIN * WW; i += 256) {
    const int ci = i / WW, w = i - ci * WW;
    plane[ci * 57 + w] = x[((size_t)(b * CIN + ci) * HH + h) * WW + w];
  }
  __syncthreads();
  // store: xpad[b][h+1][w+1][ci], ci-contiguous
  _Float16* dst = xpad + ((size_t)(b * HP + h + 1) * HP + 1) * CIN;
  for (int i = t; i < WW * CIN; i += 256) {
    const int w = i >> 8, ci = i & 255;
    dst[(size_t)w * CIN + ci] = (_Float16)plane[ci * 57 + w];
  }
}

// ---------------- Phase 6: implicit-GEMM conv, pure-register WMMA -----------------
// Block tile M=128 (cout) x N=128 (pixels); 8 waves in 4(M) x 2(N); each wave owns
// 32x64 C = 2x4 tiles. K loop: 9 taps x 8 cin-chunks of 32. A and B fragments are
// loaded straight from global (cin-contiguous layouts) as global_load_b128 pairs.
__global__ __launch_bounds__(256) void conv_kernel(const _Float16* __restrict__ xpad,
                                                   const _Float16* __restrict__ aggwT,
                                                   const float* __restrict__ aggb,
                                                   float* __restrict__ out) {
  const int b   = blockIdx.z;
  const int co0 = blockIdx.y * 128;
  const int px0 = blockIdx.x * 128;
  const int lane = threadIdx.x & 31;
  const int wm   = (threadIdx.x >> 5) >> 1;   // 0..3
  const int wn   = (threadIdx.x >> 5) & 1;    // 0..1

  const int arow = lane & 15;
  const int akb  = (lane < 16) ? 0 : 8;    // A frag: K kb..kb+7 and kb+16..kb+23
  const int bko  = (lane < 16) ? 0 : 8;    // B frag: K kb..kb+7 and kb+16..kb+23

  // Hoisted per-lane base pointers
  const _Float16* bptr[4];
#pragma unroll
  for (int nt = 0; nt < 4; ++nt) {
    int p = px0 + wn * 64 + nt * 16 + arow;
    if (p >= HW) p = 0;                  // clamp: loads valid garbage, store masked
    const int oh = p / WW, ow = p - oh * WW;
    bptr[nt] = xpad + ((size_t)(b * HP + oh) * HP + ow) * CIN + bko;
  }
  const _Float16* aptr[2];
#pragma unroll
  for (int mt = 0; mt < 2; ++mt) {
    const int m = co0 + wm * 32 + mt * 16 + arow;
    aptr[mt] = aggwT + ((size_t)(b * 9) * COUT + m) * CIN + akb;
  }

  v8f acc[2][4] = {};

  for (int tap = 0; tap < 9; ++tap) {
    const size_t boff = ((size_t)(tap / 3) * HP + (tap % 3)) * CIN;  // pad offset built in
    const size_t aoff = (size_t)tap * COUT * CIN;
    for (int cc = 0; cc < 8; ++cc) {
      const int ci0 = cc * 32;
      v16h af[2];
#pragma unroll
      for (int mt = 0; mt < 2; ++mt) {
        const _Float16* s = aptr[mt] + aoff + ci0;
        union { v16h v; v8h h[2]; } u;
        u.h[0] = *reinterpret_cast<const v8h*>(s);
        u.h[1] = *reinterpret_cast<const v8h*>(s + 16);
        af[mt] = u.v;
        if (cc < 7) __builtin_prefetch(s + 32, 0, 1);   // next cin chunk
      }
#pragma unroll
      for (int nt = 0; nt < 4; ++nt) {
        const _Float16* s = bptr[nt] + boff + ci0;
        union { v16h v; v8h h[2]; } u;
        u.h[0] = *reinterpret_cast<const v8h*>(s);
        u.h[1] = *reinterpret_cast<const v8h*>(s + 16);
        const v16h bf = u.v;
#pragma unroll
        for (int mt = 0; mt < 2; ++mt) {
          acc[mt][nt] = __builtin_amdgcn_wmma_f32_16x16x32_f16(
              false, af[mt], false, bf, (short)0, acc[mt][nt], false, false);
        }
      }
    }
  }

  // Epilogue: C layout (VGPR r: lanes0-15 M=r, lanes16-31 M=8+r; N=lane&15)
  const int mhalf = (lane < 16) ? 0 : 8;
  const int nn = lane & 15;
#pragma unroll
  for (int mt = 0; mt < 2; ++mt) {
#pragma unroll
    for (int nt = 0; nt < 4; ++nt) {
      const int p = px0 + wn * 64 + nt * 16 + nn;
      if (p < HW) {
#pragma unroll
        for (int r = 0; r < 8; ++r) {
          const int co = co0 + wm * 32 + mt * 16 + mhalf + r;
          out[((size_t)(b * COUT + co)) * HW + p] = acc[mt][nt][r] + aggb[b * COUT + co];
        }
      }
    }
  }
}

extern "C" void kernel_launch(void* const* d_in, const int* in_sizes, int n_in,
                              void* d_out, int out_size, void* d_ws, size_t ws_size,
                              hipStream_t stream) {
  const float* x     = (const float*)d_in[0];
  const float* wgt   = (const float*)d_in[1];
  const float* bias  = (const float*)d_in[2];
  const float* fc1_w = (const float*)d_in[3];
  const float* fc1_b = (const float*)d_in[4];
  const float* fc2_w = (const float*)d_in[5];
  const float* fc2_b = (const float*)d_in[6];
  float* out = (float*)d_out;

  // workspace layout
  float* att    = (float*)d_ws;                                   // 128 f32
  float* pooled = att + 128;                                      // 8192 f32
  float* aggb   = pooled + B_ * CIN;                              // 8192 f32
  _Float16* xpad  = (_Float16*)((char*)d_ws + 66560);             // 32*58*58*256 f16 = 55,115,776 B
  _Float16* aggwT = (_Float16*)((char*)d_ws + 66560 + 55115776);  // 32*9*256*256 f16 = 37,748,736 B

  const int xpad_halfs = B_ * HP * HP * CIN;                      // 27,557,888

  pool_kernel<<<B_ * CIN, 256, 0, stream>>>(x, pooled);
  attention_kernel<<<1, 128, 0, stream>>>(pooled, fc1_w, fc1_b, fc2_w, fc2_b, att);
  aggw_kernel<<<dim3(9 * COUT * CIN / 256, B_), 256, 0, stream>>>(wgt, att, aggwT);
  aggb_kernel<<<B_, COUT, 0, stream>>>(bias, att, aggb);
  zero_kernel<<<4096, 256, 0, stream>>>((v8h*)xpad, xpad_halfs / 8);
  pad_transpose_kernel<<<dim3(HH, B_), 256, 0, stream>>>(x, xpad);
  conv_kernel<<<dim3((HW + 127) / 128, COUT / 128, B_), 256, 0, stream>>>(xpad, aggwT, aggb, out);
}
